// Yolov1Model_51711406244493
// MI455X (gfx1250) — compile-verified, matrix operand order checked by reading, not run on hardware
//
#include <hip/hip_runtime.h>
#include <hip/hip_bf16.h>
#include <math.h>

// ---------------------------------------------------------------------------
// YOLOv1 head for MI455X (gfx1250, wave32).
// GEMM work (4 convs as implicit im2col, 2 FCs) on v_wmma_f32_16x16x32_bf16,
// f32 accumulation; fp32 -> bf16 via hardware v_cvt_pk_bf16_f32 while staging
// into LDS. Conv1 uses a 64x128 block tile (8 WMMAs per wave from 6 fragment
// loads); convs 2-4 use 64x64. Software-pipelined K loop, one barrier/step.
// Branchless gathers: clamped addresses + selects, no exec divergence.
// ---------------------------------------------------------------------------

typedef __attribute__((ext_vector_type(16))) __bf16 v16bf;
typedef __attribute__((ext_vector_type(8)))  __bf16 v8bf;
typedef __attribute__((ext_vector_type(4)))  __bf16 v4bf;
typedef __attribute__((ext_vector_type(8)))  float  v8f;
typedef __attribute__((ext_vector_type(2)))  float  f32x2;

__device__ __forceinline__ __bf16 to_bf16(float f) { return (__bf16)f; }  // RNE
__device__ __forceinline__ int imin(int a, int b) { return a < b ? a : b; }
__device__ __forceinline__ int imax(int a, int b) { return a > b ? a : b; }

// 16x32 bf16 fragment from LDS tile [16 rows][stride], rows k-contiguous.
// ISA layout: lane -> row lane&15, K-half 8*(lane>>4); element e ->
// k = kh + (e&7) + (e>=8?16:0)  => exactly two ds_load_b128 per fragment.
__device__ __forceinline__ v16bf frag_nk(const __bf16* base, int stride) {
  const int lane = threadIdx.x & 31;
  const int r    = lane & 15;
  const int kh   = (lane >> 4) << 3;
  const __bf16* p = base + r * stride + kh;
  v8bf lo = *(const v8bf*)(p);
  v8bf hi = *(const v8bf*)(p + 16);
  return __builtin_shufflevector(lo, hi, 0,1,2,3,4,5,6,7,8,9,10,11,12,13,14,15);
}

__device__ __forceinline__ v8f wmma_bf16(v16bf a, v16bf b, v8f c) {
  return __builtin_amdgcn_wmma_f32_16x16x32_bf16(
      false, a, false, b, (short)0, c, false, false);
}

__device__ __forceinline__ float lrelu(float v) { return v >= 0.f ? v : 0.1f * v; }

#define BM 64
#define BK 32
#define ASTR 40   // padded LDS stride (bf16): 80B rows -> conflict-free b128
#define BSTR 40

// ---------------------------------------------------------------------------
// Conv 3x3 (pad 1) as implicit-im2col GEMM: M=Cout, N=B*Ho*Wo, K=Cin*9.
// 128 threads = 4 waves. Block tile 64 x BNT.
//   BNT=128: wave tile 32x64 -> 8 WMMAs / 6 frag loads (conv1)
//   BNT= 64: wave tile 32x32 -> 4 WMMAs / 4 frag loads (convs 2-4)
// Requires K % 64 == 0 (true: 18432, 9216).
// ---------------------------------------------------------------------------
template <int BNT>
__global__ __launch_bounds__(128)
void conv3x3_wmma(const float* __restrict__ in, const float* __restrict__ w,
                  const float* __restrict__ bias, float* __restrict__ out,
                  int Bn, int Cin, int H, int W, int Cout, int Ho, int Wo,
                  int stride)
{
  constexpr int WN   = BNT / 2;       // wave tile width
  constexpr int WN16 = WN / 16;       // B fragments per wave
  constexpr int KEL  = (BNT == 128) ? 32 : 16;  // gathered k's per thread

  __shared__ __align__(16) __bf16 tA[2][BM * ASTR];   // weights [m][k]
  __shared__ __align__(16) __bf16 tB[2][BNT * BSTR];  // im2col  [n][k]

  const int tid  = threadIdx.x;
  const int lane = tid & 31;
  const int wave = tid >> 5;
  const int K    = Cin * 9;
  const int HoWo = Ho * Wo;
  const int Ntot = Bn * HoWo;
  const int m0   = blockIdx.y * BM;
  const int n0   = blockIdx.x * BNT;
  const int wm   = (wave >> 1) * 32;
  const int wn   = (wave & 1) * WN;

  // Loop-invariant staging coordinates.
  const int aml = tid >> 3;          // A row base (+16 per e)
  const int akq = (tid & 7) * 4;     // A quad column (float4)
  const int nl  = (BNT == 128) ? tid : (tid & 63);   // B tile column
  const int kc  = (BNT == 128) ? 0 : ((tid >> 6) * 16);
  // Clamp n for addressing; out-of-range columns produce garbage that stays
  // in WMMA columns the epilogue never writes.
  const int ncl = imin(n0 + nl, Ntot - 1);
  const int pb  = ncl / HoWo;
  const int q0  = ncl - pb * HoWo;
  const int poy = q0 / Wo;
  const int pox = q0 - poy * Wo;
  const int iy0 = poy * stride - 1;
  const int ix0 = pox * stride - 1;
  const float* inb_base = in + (size_t)pb * Cin * H * W;

  v8f acc[2][WN16];
#pragma unroll
  for (int i = 0; i < 2; ++i)
#pragma unroll
    for (int j = 0; j < WN16; ++j) acc[i][j] = (v8f){};

  float4 ra[4];
  float  rb[KEL];

  auto load_stage = [&](int k0) {
#pragma unroll
    for (int e = 0; e < 4; ++e)
      ra[e] = *(const float4*)(w + (size_t)(m0 + aml + e * 16) * K + k0 + akq);
    __builtin_prefetch(w + (size_t)(m0 + aml) * K + k0 + BK, 0, 1);
#pragma unroll
    for (int e = 0; e < KEL; ++e) {
      int k   = k0 + kc + e;         // thread owns KEL contiguous k
      int cin = k / 9;
      int rem = k - cin * 9;
      int ry  = rem / 3;
      int rx  = rem - ry * 3;
      int iy  = iy0 + ry;
      int ix  = ix0 + rx;
      bool inb = ((unsigned)iy < (unsigned)H) && ((unsigned)ix < (unsigned)W);
      int iyc = imin(imax(iy, 0), H - 1);
      int ixc = imin(imax(ix, 0), W - 1);
      float v = inb_base[((size_t)cin * H + iyc) * W + ixc];  // always load
      rb[e] = inb ? v : 0.f;                                   // v_cndmask
    }
  };
  auto store_stage = [&](int p) {
    __bf16* A  = tA[p];
    __bf16* Bt = tB[p];
#pragma unroll
    for (int e = 0; e < 4; ++e) {
      v4bf d = { to_bf16(ra[e].x), to_bf16(ra[e].y),
                 to_bf16(ra[e].z), to_bf16(ra[e].w) };
      *(v4bf*)(A + (aml + e * 16) * ASTR + akq) = d;      // ds_store_b64
    }
#pragma unroll
    for (int g = 0; g < KEL / 8; ++g) {
      v8bf d = { to_bf16(rb[g*8+0]), to_bf16(rb[g*8+1]),
                 to_bf16(rb[g*8+2]), to_bf16(rb[g*8+3]),
                 to_bf16(rb[g*8+4]), to_bf16(rb[g*8+5]),
                 to_bf16(rb[g*8+6]), to_bf16(rb[g*8+7]) };
      *(v8bf*)(Bt + nl * BSTR + kc + g * 8) = d;          // ds_store_b128
    }
  };
  auto compute = [&](int p) {
    const __bf16* A  = tA[p];
    const __bf16* Bt = tB[p];
    v16bf a0 = frag_nk(A + (wm +  0) * ASTR, ASTR);
    v16bf a1 = frag_nk(A + (wm + 16) * ASTR, ASTR);
#pragma unroll
    for (int j = 0; j < WN16; ++j) {
      v16bf bj = frag_nk(Bt + (wn + j * 16) * BSTR, BSTR);
      acc[0][j] = wmma_bf16(a0, bj, acc[0][j]);
      acc[1][j] = wmma_bf16(a1, bj, acc[1][j]);
    }
  };

  load_stage(0);
  store_stage(0);
  __syncthreads();
  int p = 0;
  for (int k0 = 0; k0 < K; k0 += BK) {
    const bool more = (k0 + BK) < K;
    if (more) load_stage(k0 + BK);   // global loads in flight...
    compute(p);                      // ...overlap the WMMAs
    if (more) store_stage(p ^ 1);
    __syncthreads();
    p ^= 1;
  }

  // Epilogue: C layout col = lane&15, row = v + 8*(lane>>4). Bias + LeakyReLU.
  const int colL = lane & 15;
  const int rowH = (lane >> 4) * 8;
#pragma unroll
  for (int j = 0; j < WN16; ++j) {
    int nn = n0 + wn + j * 16 + colL;
    if (nn >= Ntot) continue;
    int b  = nn / HoWo;
    int q  = nn - b * HoWo;
    int oy = q / Wo;
    int ox = q - oy * Wo;
#pragma unroll
    for (int i = 0; i < 2; ++i) {
#pragma unroll
      for (int v = 0; v < 8; ++v) {
        int m = m0 + wm + i * 16 + rowH + v;
        out[(((size_t)b * Cout + m) * Ho + oy) * Wo + ox] =
            lrelu(acc[i][j][v] + bias[m]);
      }
    }
  }
}

// ---------------------------------------------------------------------------
// FC layer, split-K: partial[z][16][Npad] = x[16, chunk] @ W[chunk, N].
// M=16 = one WMMA row tile; 4 waves cover 64 columns. Requires
// kchunk % 64 == 0 (true: 6272, 512).
// ---------------------------------------------------------------------------
#define BN 64

__global__ __launch_bounds__(128)
void fc_wmma_splitk(const float* __restrict__ x, const float* __restrict__ wgt,
                    float* __restrict__ partial, int K, int N, int Npad,
                    int kchunk)
{
  __shared__ __align__(16) __bf16 tA[2][16 * ASTR];
  __shared__ __align__(16) __bf16 tB[2][BN * BSTR];

  const int tid  = threadIdx.x;
  const int lane = tid & 31;
  const int wave = tid >> 5;
  const int n0   = blockIdx.x * BN;
  const int kbeg = blockIdx.z * kchunk;
  const int kend = kbeg + kchunk;

  const int aml = tid >> 3;          // A row (0..15)
  const int akq = (tid & 7) * 4;     // A quad column
  const int np2 = (tid & 31) * 2;    // B column pair (invariant)
  const int bk8 = (tid >> 5) * 8;    // B k-quarter base (0,8,16,24)
  // Clamp tail columns (N even): garbage lands in columns never written.
  const int nsafe = imin(n0 + np2, N - 2);

  v8f acc = {};
  float4 ra;
  f32x2  rbv[8];

  auto load_stage = [&](int k0) {
    ra = *(const float4*)(x + (size_t)aml * K + k0 + akq);
#pragma unroll
    for (int e = 0; e < 8; ++e)
      rbv[e] = *(const f32x2*)(wgt + (size_t)(k0 + bk8 + e) * N + nsafe);
    __builtin_prefetch(wgt + (size_t)(k0 + BK + bk8) * N + nsafe, 0, 0);
  };
  auto store_stage = [&](int p) {
    __bf16* A  = tA[p];
    __bf16* Bt = tB[p];
    v4bf d = { to_bf16(ra.x), to_bf16(ra.y), to_bf16(ra.z), to_bf16(ra.w) };
    *(v4bf*)(A + aml * ASTR + akq) = d;                   // ds_store_b64
    v8bf cx = { to_bf16(rbv[0].x), to_bf16(rbv[1].x), to_bf16(rbv[2].x),
                to_bf16(rbv[3].x), to_bf16(rbv[4].x), to_bf16(rbv[5].x),
                to_bf16(rbv[6].x), to_bf16(rbv[7].x) };
    v8bf cy = { to_bf16(rbv[0].y), to_bf16(rbv[1].y), to_bf16(rbv[2].y),
                to_bf16(rbv[3].y), to_bf16(rbv[4].y), to_bf16(rbv[5].y),
                to_bf16(rbv[6].y), to_bf16(rbv[7].y) };
    *(v8bf*)(Bt + (np2    ) * BSTR + bk8) = cx;           // ds_store_b128
    *(v8bf*)(Bt + (np2 + 1) * BSTR + bk8) = cy;           // ds_store_b128
  };
  auto compute = [&](int p) {
    v16bf a = frag_nk(tA[p], ASTR);
    v16bf b = frag_nk(tB[p] + (wave * 16) * BSTR, BSTR);
    acc = wmma_bf16(a, b, acc);
  };

  load_stage(kbeg);
  store_stage(0);
  __syncthreads();
  int p = 0;
  for (int k0 = kbeg; k0 < kend; k0 += BK) {
    const bool more = (k0 + BK) < kend;
    if (more) load_stage(k0 + BK);
    compute(p);
    if (more) store_stage(p ^ 1);
    __syncthreads();
    p ^= 1;
  }

  const int col = n0 + wave * 16 + (lane & 15);
  if (col < N) {
    const int rowH = (lane >> 4) * 8;
#pragma unroll
    for (int v = 0; v < 8; ++v)
      partial[(size_t)(blockIdx.z * 16 + rowH + v) * Npad + col] = acc[v];
  }
}

// Reduce split-K partials, add bias, apply activation (0=lrelu, 1=sigmoid).
__global__ void fc_reduce(const float* __restrict__ partial,
                          const float* __restrict__ bias,
                          float* __restrict__ out,
                          int N, int Npad, int ksplit, int act)
{
  int t = blockIdx.x * blockDim.x + threadIdx.x;
  if (t >= 16 * N) return;
  int m = t / N;
  int n = t - m * N;
  float s = 0.f;
  for (int z = 0; z < ksplit; ++z)
    s += partial[(size_t)(z * 16 + m) * Npad + n];
  s += bias[n];
  s = (act == 0) ? lrelu(s) : 1.f / (1.f + expf(-s));
  out[(size_t)m * N + n] = s;
}

// ---------------------------------------------------------------------------
// Decode + NMS + outputs. One block per image; 98 boxes, greedy NMS matching
// the reference (argsort(-score), kept boxes suppress strictly-later ones).
// ---------------------------------------------------------------------------
#define NBOX 98

__global__ __launch_bounds__(128)
void decode_nms(const float* __restrict__ pred,
                float* __restrict__ boxes_out, float* __restrict__ scores_out,
                float* __restrict__ keep_out, float* __restrict__ labels_out)
{
  __shared__ float bx1[NBOX], by1[NBOX], bx2[NBOX], by2[NBOX], sc[NBOX], s2[NBOX];
  __shared__ int ord[NBOX];
  __shared__ unsigned char keep[NBOX], sup[NBOX];

  const int b = blockIdx.x;
  const int t = threadIdx.x;
  const float cell = 1.f / 7.f;
  const float* p = pred + (size_t)b * 1470;

  if (t < NBOX) {
    // n = (i*7 + j)*2 + nb ; o[b,i,j,:] = out[b,j,i,:]
    int i   = t / 14;
    int rem = t - i * 14;
    int j   = rem >> 1;
    int nb  = rem & 1;
    int base = (j * 7 + i) * 30 + nb * 5;
    float tx = p[base + 0], ty = p[base + 1];
    float tw = p[base + 2], th = p[base + 3], s = p[base + 4];
    float cx = (tx + (float)i) * cell;
    float cy = (ty + (float)j) * cell;
    bx1[t] = cx - 0.5f * tw;  by1[t] = cy - 0.5f * th;
    bx2[t] = cx + 0.5f * tw;  by2[t] = cy + 0.5f * th;
    sc[t]  = s;
    s2[t]  = (s >= 0.2f) ? s : -1.f;
    keep[t] = 0; sup[t] = 0;
  }
  __syncthreads();

  if (t == 0) {
    for (int a = 0; a < NBOX; ++a) ord[a] = a;
    for (int a = 0; a < NBOX; ++a) {        // selection sort, score desc
      int best = a;
      for (int c = a + 1; c < NBOX; ++c)
        if (s2[ord[c]] > s2[ord[best]]) best = c;
      int tmp = ord[a]; ord[a] = ord[best]; ord[best] = tmp;
    }
    for (int a = 0; a < NBOX; ++a) {        // greedy NMS
      int ia = ord[a];
      if (s2[ia] < 0.f || sup[ia]) continue;
      keep[ia] = 1;
      float areaA = (bx2[ia] - bx1[ia]) * (by2[ia] - by1[ia]);
      for (int c = a + 1; c < NBOX; ++c) {
        int ic = ord[c];
        float iw = fminf(bx2[ia], bx2[ic]) - fmaxf(bx1[ia], bx1[ic]);
        float ih = fminf(by2[ia], by2[ic]) - fmaxf(by1[ia], by1[ic]);
        iw = fmaxf(iw, 0.f); ih = fmaxf(ih, 0.f);
        float inter = iw * ih;
        float areaC = (bx2[ic] - bx1[ic]) * (by2[ic] - by1[ic]);
        float iou = inter / (areaA + areaC - inter);
        if (iou > 0.5f) sup[ic] = 1;
      }
    }
  }
  __syncthreads();

  if (t < NBOX) {
    int ob = b * NBOX + t;
    float* bp = boxes_out + (size_t)ob * 4;
    bp[0] = keep[t] ? bx1[t] * 448.f : 0.f;
    bp[1] = keep[t] ? by1[t] * 448.f : 0.f;
    bp[2] = keep[t] ? bx2[t] * 448.f : 0.f;
    bp[3] = keep[t] ? by2[t] * 448.f : 0.f;
    scores_out[ob] = keep[t] ? sc[t] : 0.f;
    keep_out[ob]   = keep[t] ? 1.f : 0.f;
  }
  // labels[b, i*7+j, c] = out[b, j, i, 10+c]
  for (int e = t; e < 49 * 20; e += 128) {
    int cellI = e / 20;
    int c     = e - cellI * 20;
    int i     = cellI / 7;
    int j     = cellI - i * 7;
    labels_out[((size_t)b * 49 + cellI) * 20 + c] = p[(j * 7 + i) * 30 + 10 + c];
  }
}

// ---------------------------------------------------------------------------
// Driver
// ---------------------------------------------------------------------------
extern "C" void kernel_launch(void* const* d_in, const int* in_sizes, int n_in,
                              void* d_out, int out_size, void* d_ws, size_t ws_size,
                              hipStream_t stream)
{
  (void)in_sizes; (void)n_in; (void)out_size; (void)ws_size;

  const float* feat = (const float*)d_in[0];
  const float* w1   = (const float*)d_in[1];
  const float* b1   = (const float*)d_in[2];
  const float* w2   = (const float*)d_in[3];
  const float* b2   = (const float*)d_in[4];
  const float* w3   = (const float*)d_in[5];
  const float* b3   = (const float*)d_in[6];
  const float* w4   = (const float*)d_in[7];
  const float* b4   = (const float*)d_in[8];
  const float* fc1w = (const float*)d_in[9];
  const float* fc1b = (const float*)d_in[10];
  const float* fc2w = (const float*)d_in[11];
  const float* fc2b = (const float*)d_in[12];

  // Workspace layout (floats), ~25.5 MB total.
  float* ws  = (float*)d_ws;
  float* x1  = ws;                            // 16*1024*196
  float* x2  = x1 + (size_t)16 * 1024 * 196;  // 16*1024*49
  float* x3  = x2 + (size_t)16 * 1024 * 49;
  float* x4  = x3 + (size_t)16 * 1024 * 49;
  float* p1  = x4 + (size_t)16 * 1024 * 49;   // 8*16*4096
  float* fo1 = p1 + (size_t)8 * 16 * 4096;    // 16*4096
  float* p2  = fo1 + (size_t)16 * 4096;       // 8*16*1472
  float* fo2 = p2 + (size_t)8 * 16 * 1472;    // 16*1470

  dim3 blk(128);

  // Conv1: N=3136 -> 25 x 16 blocks of 64x128 (wide tiles halve w1 re-reads).
  conv3x3_wmma<128><<<dim3(25, 16), blk, 0, stream>>>(feat, w1, b1, x1,
      16, 2048, 14, 14, 1024, 14, 14, 1);
  // Convs 2-4: N=784 -> 13 x 16 blocks of 64x64 (keep block count high).
  conv3x3_wmma<64><<<dim3(13, 16), blk, 0, stream>>>(x1, w2, b2, x2,
      16, 1024, 14, 14, 1024, 7, 7, 2);
  conv3x3_wmma<64><<<dim3(13, 16), blk, 0, stream>>>(x2, w3, b3, x3,
      16, 1024, 7, 7, 1024, 7, 7, 1);
  conv3x3_wmma<64><<<dim3(13, 16), blk, 0, stream>>>(x3, w4, b4, x4,
      16, 1024, 7, 7, 1024, 7, 7, 1);

  // FC1: split-K=8 (6272 each) to spread the 822MB weight stream over 512
  // blocks; deterministic partial-sum reduction (no atomics).
  fc_wmma_splitk<<<dim3(64, 1, 8), blk, 0, stream>>>(x4, fc1w, p1,
      50176, 4096, 4096, 6272);
  fc_reduce<<<dim3(256), 256, 0, stream>>>(p1, fc1b, fo1, 4096, 4096, 8, 0);

  fc_wmma_splitk<<<dim3(23, 1, 8), blk, 0, stream>>>(fo1, fc2w, p2,
      4096, 1470, 1472, 512);
  fc_reduce<<<dim3((16 * 1470 + 255) / 256), 256, 0, stream>>>(
      p2, fc2b, fo2, 1470, 1472, 8, 1);

  float* boxes_o  = (float*)d_out;
  float* scores_o = boxes_o + (size_t)16 * 98 * 4;
  float* keep_o   = scores_o + (size_t)16 * 98;
  float* labels_o = keep_o + (size_t)16 * 98;
  decode_nms<<<dim3(16), blk, 0, stream>>>(fo2, boxes_o, scores_o, keep_o, labels_o);
}
